// CodeBook_12841952215571
// MI455X (gfx1250) — matrix operand order repaired and checked by reading, not compiled, not run on hardware
//
#include <hip/hip_runtime.h>
#include <stdint.h>

// Problem constants (match reference)
#define BATCH   65536
#define D_IN    512
#define D_OUT   512
#define NKEYS   1024

// Tiling
#define M_TILE  128                // x rows per block (8 waves x 16 rows)
#define NCHUNK  16                 // keys per chunk (one WMMA N-tile)
#define NITER   (NKEYS / NCHUNK)   // 64
#define XS      520                // padded LDS row stride (bf16): 512 + 8 -> conflict-free

typedef __attribute__((ext_vector_type(16))) __bf16 v16bf;
typedef __attribute__((ext_vector_type(8)))  float  v8f;

union ABf { uint4 u4[2]; v16bf v; };

static __device__ __forceinline__ uint32_t bf16pair(float a, float b) {
  uint32_t ua = __float_as_uint(a);
  uint32_t ub = __float_as_uint(b);
  ua = (ua + 0x7FFFu + ((ua >> 16) & 1u)) >> 16;   // RNE
  ub = (ub + 0x7FFFu + ((ub >> 16) & 1u)) >> 16;
  return ua | (ub << 16);
}

// ---------------------------------------------------------------------------
// Prep: keys f32 -> bf16 (packed u32 pairs) + 0.5*|k|^2 per key
// grid = NKEYS blocks, 256 threads; one key row per block
// ---------------------------------------------------------------------------
__global__ __launch_bounds__(256)
void codebook_prep_keys(const float* __restrict__ keys,
                        uint32_t* __restrict__ kb,      // [NKEYS][D_IN/2] u32
                        float* __restrict__ k2h) {      // [NKEYS]
  const int row = blockIdx.x;
  const int t   = threadIdx.x;
  const float2 v = ((const float2*)(keys + (size_t)row * D_IN))[t];
  kb[(size_t)row * (D_IN / 2) + t] = bf16pair(v.x, v.y);

  float s = v.x * v.x + v.y * v.y;
  #pragma unroll
  for (int off = 16; off > 0; off >>= 1) s += __shfl_xor(s, off, 32);

  __shared__ float wsum[8];
  const int wv = t >> 5, lane = t & 31;
  if (lane == 0) wsum[wv] = s;
  __syncthreads();
  if (t == 0) {
    float tot = 0.f;
    #pragma unroll
    for (int i = 0; i < 8; ++i) tot += wsum[i];
    k2h[row] = 0.5f * tot;
  }
}

// ---------------------------------------------------------------------------
// Main fused kernel: bf16 WMMA GEMM scores + running argmax + values gather
// grid = BATCH/M_TILE blocks, 256 threads (8 wave32)
// ---------------------------------------------------------------------------
__global__ __launch_bounds__(256)
void codebook_main(const float* __restrict__ x,
                   const uint4* __restrict__ kb4,     // keys bf16: 64 uint4 per key row
                   const float* __restrict__ k2h,
                   const float* __restrict__ values,
                   float* __restrict__ out) {
  extern __shared__ char smem[];
  uint16_t* xT     = (uint16_t*)smem;                 // [M_TILE][XS] bf16
  uint16_t* kT     = xT + M_TILE * XS;                // [2][NCHUNK][XS] bf16 (double buf)
  int*      idxRow = (int*)(kT + 2 * NCHUNK * XS);    // [M_TILE]

  const int t  = threadIdx.x;
  const int b0 = blockIdx.x * M_TILE;

  // ---- Stage X tile (f32 -> bf16) into LDS: 128 rows x 512, float4 coalesced
  {
    const float4* x4 = (const float4*)x + (size_t)b0 * (D_IN / 4);
    #pragma unroll
    for (int i = 0; i < (M_TILE * D_IN / 4) / 256; ++i) {   // 64 float4 / thread
      const int li  = t + i * 256;
      const int row = li >> 7;          // 128 float4 per row
      const int col = li & 127;
      const float4 v = x4[(size_t)row * 128 + col];
      uint2 p;
      p.x = bf16pair(v.x, v.y);
      p.y = bf16pair(v.z, v.w);
      *(uint2*)(xT + row * XS + col * 4) = p;
    }
  }

  // ---- Keys chunk streaming: 4 named uint4 regs (no array -> no scratch spill)
  // Global side: chunk nc occupies kb4[nc*1024 .. nc*1024+1023], thread t takes
  // elements {t, t+256, t+512, t+768} (coalesced).
  // LDS side: li = t + i*256 -> row = (t>>6) + 4*i, col = t&63 within the chunk.
  uint16_t* kst = kT + (t >> 6) * XS + (t & 63) * 8;  // per-thread LDS store base
  uint4 kr0, kr1, kr2, kr3;

  #define LOAD_KCHUNK(nc)                                    \
    { const uint4* g = kb4 + (size_t)(nc) * 1024 + t;        \
      kr0 = g[0]; kr1 = g[256]; kr2 = g[512]; kr3 = g[768]; }
  #define STORE_KCHUNK(buf)                                  \
    { uint16_t* s = kst + (buf) * (NCHUNK * XS);             \
      *(uint4*)(s)           = kr0;                          \
      *(uint4*)(s + 4 * XS)  = kr1;                          \
      *(uint4*)(s + 8 * XS)  = kr2;                          \
      *(uint4*)(s + 12 * XS) = kr3; }

  LOAD_KCHUNK(0)
  STORE_KCHUNK(0)
  __syncthreads();

  const int wv    = t >> 5;        // wave id: M sub-tile
  const int lane  = t & 31;
  const int l15   = lane & 15;
  const int half8 = (lane >> 4) * 8;
  const uint16_t* xrow = xT + (wv * 16 + l15) * XS;

  float bestV[8];
  int   bestI[8];
  #pragma unroll
  for (int j = 0; j < 8; ++j) { bestV[j] = -3.0e38f; bestI[j] = 0; }

  // A frag (16x32 bf16, ISA 7.12.2): V0-3: K=half*8+0..7, V4-7: K=16+half*8+0..7
  // B frag (32x16 bf16): lane = key column, same K layout as A
  #define LDA(frag, d0)                                          \
    frag.u4[0] = *(const uint4*)(xrow + (d0) + half8);           \
    frag.u4[1] = *(const uint4*)(xrow + (d0) + 16 + half8);
  #define LDB(frag, d0)                                          \
    frag.u4[0] = *(const uint4*)(krow + (d0));                   \
    frag.u4[1] = *(const uint4*)(krow + (d0) + 16);

  for (int nc = 0; nc < NITER; ++nc) {
    const int cur = nc & 1;
    if (nc + 1 < NITER) LOAD_KCHUNK(nc + 1)      // overlap global latency w/ WMMA

    const uint16_t* krow = kT + cur * (NCHUNK * XS) + l15 * XS + half8;

    v8f acc = {};
    ABf a0, b0, a1, b1;
    LDA(a0, 0) LDB(b0, 0)
    #pragma unroll
    for (int ds2 = 0; ds2 < 16; ds2 += 2) {      // K = 512 = 16 x 32, 2-deep pipeline
      LDA(a1, (ds2 + 1) * 32) LDB(b1, (ds2 + 1) * 32)
      acc = __builtin_amdgcn_wmma_f32_16x16x32_bf16(false, a0.v, false, b0.v,
                                                    (short)0, acc, false, false);
      if (ds2 + 2 < 16) { LDA(a0, (ds2 + 2) * 32) LDB(b0, (ds2 + 2) * 32) }
      acc = __builtin_amdgcn_wmma_f32_16x16x32_bf16(false, a1.v, false, b1.v,
                                                    (short)0, acc, false, false);
    }

    // score = x.k - 0.5|k|^2 ; argmin(dist) == argmax(score), first index on ties
    const float kk   = k2h[nc * NCHUNK + l15];
    const int   kidx = nc * NCHUNK + l15;
    #pragma unroll
    for (int j = 0; j < 8; ++j) {                // C layout: VGPR j -> rows j / j+8
      const float v = acc[j] - kk;
      if (v > bestV[j]) { bestV[j] = v; bestI[j] = kidx; }
    }

    if (nc + 1 < NITER) STORE_KCHUNK(cur ^ 1)
    __syncthreads();
  }

  #undef LDA
  #undef LDB
  #undef LOAD_KCHUNK
  #undef STORE_KCHUNK

  // ---- Cross-lane argmax within each 16-lane half (rows wv*16+half*8+j)
  #pragma unroll
  for (int j = 0; j < 8; ++j) {
    #pragma unroll
    for (int off = 1; off < 16; off <<= 1) {
      const float ov = __shfl_xor(bestV[j], off, 32);
      const int   oi = __shfl_xor(bestI[j], off, 32);
      if (ov > bestV[j] || (ov == bestV[j] && oi < bestI[j])) {
        bestV[j] = ov; bestI[j] = oi;
      }
    }
    if (l15 == 0) idxRow[wv * 16 + half8 + j] = bestI[j];
  }
  __syncthreads();

  // ---- Gather values[argmin] rows -> out (float4, coalesced; values L2-hot)
  const float4* val4 = (const float4*)values;
  float4*       out4 = (float4*)out + (size_t)b0 * (D_OUT / 4);
  #pragma unroll
  for (int i = 0; i < (M_TILE * D_OUT / 4) / 256; ++i) {
    const int li  = t + i * 256;
    const int row = li >> 7;
    const int col = li & 127;
    out4[(size_t)row * 128 + col] = val4[(size_t)idxRow[row] * 128 + col];
  }
}

// ---------------------------------------------------------------------------
extern "C" void kernel_launch(void* const* d_in, const int* in_sizes, int n_in,
                              void* d_out, int out_size, void* d_ws, size_t ws_size,
                              hipStream_t stream) {
  const float* x      = (const float*)d_in[0];
  const float* keys   = (const float*)d_in[1];
  const float* values = (const float*)d_in[2];
  float* out = (float*)d_out;

  // workspace: keys bf16 (1 MB) then 0.5*|k|^2 (4 KB)
  uint32_t* kb  = (uint32_t*)d_ws;
  float*    k2h = (float*)((char*)d_ws + (size_t)NKEYS * D_IN * 2);

  codebook_prep_keys<<<NKEYS, 256, 0, stream>>>(keys, kb, k2h);

  const size_t smem_bytes = (size_t)(M_TILE * XS + 2 * NCHUNK * XS) * 2 + M_TILE * 4;
  codebook_main<<<BATCH / M_TILE, 256, smem_bytes, stream>>>(
      x, (const uint4*)kb, k2h, values, out);
}
